// MonotonicBahaAtt_45105746542943
// MI455X (gfx1250) — matrix operand order repaired and checked by reading, not compiled, not run on hardware
//
#include <hip/hip_runtime.h>

// ---------------------------------------------------------------------------
// Problem constants (B=64, S=2048, D=U=1024)
// ---------------------------------------------------------------------------
#define BATCH 64
#define SEQ   2048
#define DIM   1024
#define UNITS 1024
#define EPS   1e-10f

#define ASTRIDE 1032   // 1024 + 8 bf16 pad -> rows land in different LDS banks

typedef __attribute__((ext_vector_type(16))) __bf16        v16bf;
typedef __attribute__((ext_vector_type(8)))  float         v8f;
typedef __attribute__((ext_vector_type(4)))  unsigned int  v4u;
typedef __attribute__((ext_vector_type(8)))  unsigned int  v8u;

__device__ __forceinline__ unsigned short f2bf(float f) {
    unsigned int u = __builtin_bit_cast(unsigned int, f);
    u += 0x7FFFu + ((u >> 16) & 1u);       // round-to-nearest-even truncation
    return (unsigned short)(u >> 16);
}

// Hardware V_TANH_F32 (TRANS op, co-executes with WMMA); ocml fallback.
#if __has_builtin(__builtin_amdgcn_tanhf)
__device__ __forceinline__ float fast_tanh(float x) { return __builtin_amdgcn_tanhf(x); }
#else
__device__ __forceinline__ float fast_tanh(float x) { return tanhf(x); }
#endif

// ---------------------------------------------------------------------------
// Kernel 0: Ua [D,U] f32 -> UaT [U,D] bf16 (transposed so each WMMA B lane
// reads 16 contiguous K values). 32x32 LDS tile transpose.
// ---------------------------------------------------------------------------
__global__ __launch_bounds__(256) void ua_transpose_bf16(
    const float* __restrict__ Ua, unsigned short* __restrict__ UaT) {
    __shared__ float tile[32][33];
    const int kt = blockIdx.x * 32;
    const int nt = blockIdx.y * 32;
    const int tx = threadIdx.x;        // 0..31
    const int ty = threadIdx.y;        // 0..7
    #pragma unroll
    for (int i = 0; i < 32; i += 8)
        tile[ty + i][tx] = Ua[(size_t)(kt + ty + i) * UNITS + nt + tx];
    __syncthreads();
    #pragma unroll
    for (int i = 0; i < 32; i += 8)
        UaT[(size_t)(nt + ty + i) * DIM + kt + tx] = f2bf(tile[tx][ty + i]);
}

// ---------------------------------------------------------------------------
// Kernel 1: dec_proj[b,n] = sum_k dec[b,k] * Wa[k,n]   (small, f32 VALU)
// ---------------------------------------------------------------------------
__global__ __launch_bounds__(256) void decproj_kernel(
    const float* __restrict__ dec, const float* __restrict__ Wa,
    float* __restrict__ out) {
    const int b = blockIdx.x;
    const int n0 = threadIdx.x;
    float acc0 = 0.f, acc1 = 0.f, acc2 = 0.f, acc3 = 0.f;
    const float* dr = dec + (size_t)b * DIM;
    for (int k = 0; k < DIM; ++k) {
        const float d = dr[k];
        const float* wr = Wa + (size_t)k * UNITS + n0;
        acc0 = fmaf(d, wr[0],   acc0);
        acc1 = fmaf(d, wr[256], acc1);
        acc2 = fmaf(d, wr[512], acc2);
        acc3 = fmaf(d, wr[768], acc3);
    }
    float* o = out + (size_t)b * UNITS + n0;
    o[0] = acc0; o[256] = acc1; o[512] = acc2; o[768] = acc3;
}

// ---------------------------------------------------------------------------
// Kernel 2: fused Bahdanau score with bf16 WMMA, software-pipelined K loop.
//   score[b,s] = sum_n tanh( enc[b,s,:]@Ua[:,n] + dec_proj[b,n] + Uab[n] ) * Va[n]
// 2048 blocks x 256 threads (8 waves): 64 (b,s) rows per block staged in LDS
// as bf16; wave w handles row-subtile (w&3)*16 and N-half (w>>2)*512.
// ---------------------------------------------------------------------------
__global__ __launch_bounds__(256) void fused_score_kernel(
    const float* __restrict__ enc, const unsigned short* __restrict__ UaT,
    const float* __restrict__ decproj, const float* __restrict__ Uab,
    const float* __restrict__ Va, float* __restrict__ score) {
    extern __shared__ unsigned char smem[];
    unsigned short* Abuf = (unsigned short*)smem;                    // 64*1032*2 = 132096 B
    float* dpb  = (float*)(smem + 64 * ASTRIDE * 2);                 // 1024 f32
    float* vab  = dpb + UNITS;                                       // 1024 f32
    float* sAcc = vab + UNITS;                                       // 64 f32

    const int t = threadIdx.x;
    const int wgRow = blockIdx.x * 64;          // flat (b*S + s) row base
    const int b = wgRow >> 11;                  // S = 2048 rows per batch

    // ---- stage A: 64 rows x 1024 f32 -> bf16 in LDS ----
    const float4* encv = (const float4*)(enc + (size_t)wgRow * DIM);
    unsigned int* AbufU = (unsigned int*)Abuf;
    #pragma unroll 4
    for (int it = 0; it < 64; ++it) {
        const int idx = t + it * 256;           // 0..16383 float4s
        const int row = idx >> 8;               // 256 float4 per row
        const int c4  = idx & 255;
        const float4 v = encv[row * 256 + c4];
        const unsigned int lo = (unsigned int)f2bf(v.x) | ((unsigned int)f2bf(v.y) << 16);
        const unsigned int hi = (unsigned int)f2bf(v.z) | ((unsigned int)f2bf(v.w) << 16);
        const int e = row * ASTRIDE + c4 * 4;   // bf16-element offset (even)
        AbufU[(e >> 1) + 0] = lo;
        AbufU[(e >> 1) + 1] = hi;
    }
    for (int i = t; i < UNITS; i += 256) {
        dpb[i] = decproj[(size_t)b * UNITS + i] + Uab[i];
        vab[i] = Va[i];
    }
    if (t < 64) sAcc[t] = 0.0f;
    __syncthreads();

    // ---- WMMA main loop ----
    const int lane  = t & 31;
    const int wave  = t >> 5;
    const int sub   = wave & 3;                 // row subtile 0..3
    const int nhalf = wave >> 2;                // N half 0..1
    const int ln    = lane & 15;
    const bool hi16 = lane >= 16;

    const unsigned short* arow = Abuf + (sub * 16 + ln) * ASTRIDE;   // A row = M
    const int kloOff = hi16 ? 16 : 0;           // A lane K layout: contiguous k = 16h + s
    const int khiOff = hi16 ? 24 : 8;
    const int kbOff  = hi16 ? 16 : 0;           // B lane K layout (32x16 bf16)

    float part[8] = {0.f, 0.f, 0.f, 0.f, 0.f, 0.f, 0.f, 0.f};

    const int ncBeg = nhalf * 32;
    for (int nc = ncBeg; nc < ncBeg + 32; ++nc) {
        const int n = nc * 16 + ln;             // this lane's N column
        const unsigned short* bcol = UaT + (size_t)n * DIM + kbOff;

        // Pull next N-chunk's B column toward L2/L0 while we crunch this one.
        if (nc + 1 < ncBeg + 32)
            __builtin_prefetch(UaT + (size_t)(n + 16) * DIM + kbOff, 0, 3);

        v8f acc = {0.f, 0.f, 0.f, 0.f, 0.f, 0.f, 0.f, 0.f};

        // Software pipeline, distance 1, FULLY unrolled: rotation registers
        // become SSA values -> no v_mov rotation / hazard nops, loads for
        // kc+1 in flight while the WMMA for kc executes (partial s_wait).
        v8u br = *(const v8u*)(bcol);
        v4u a0 = *(const v4u*)(arow + kloOff);
        v4u a1 = *(const v4u*)(arow + khiOff);
        #pragma unroll
        for (int kc = 0; kc < 32; ++kc) {
            v8u brn; v4u a0n, a1n;
            if (kc < 31) {
                const int kbn = (kc + 1) * 32;
                brn = *(const v8u*)(bcol + kbn);
                a0n = *(const v4u*)(arow + kbn + kloOff);
                a1n = *(const v4u*)(arow + kbn + khiOff);
            } else {
                brn = br; a0n = a0; a1n = a1;
            }
            v8u ar;
            ar[0] = a0[0]; ar[1] = a0[1]; ar[2] = a0[2]; ar[3] = a0[3];
            ar[4] = a1[0]; ar[5] = a1[1]; ar[6] = a1[2]; ar[7] = a1[3];
            acc = __builtin_amdgcn_wmma_f32_16x16x32_bf16(
                false, __builtin_bit_cast(v16bf, ar),
                false, __builtin_bit_cast(v16bf, br),
                (short)0, acc, false, false);
            br = brn; a0 = a0n; a1 = a1n;
        }

        const float dpn = dpb[n];
        const float van = vab[n];
        #pragma unroll
        for (int i = 0; i < 8; ++i)
            part[i] = fmaf(fast_tanh(acc[i] + dpn), van, part[i]);
    }

    // ---- reduce over the 16 lanes holding each row's N columns ----
    #pragma unroll
    for (int i = 0; i < 8; ++i) {
        float v = part[i];
        v += __shfl_xor(v, 1, 32);
        v += __shfl_xor(v, 2, 32);
        v += __shfl_xor(v, 4, 32);
        v += __shfl_xor(v, 8, 32);
        part[i] = v;
    }
    if (ln == 0) {                               // lanes 0 and 16
        const int rbase = sub * 16 + (hi16 ? 8 : 0);   // C/D layout: M=i / M=i+8
        #pragma unroll
        for (int i = 0; i < 8; ++i)
            atomicAdd(&sAcc[rbase + i], part[i]);      // 2 adds/slot: deterministic
    }
    __syncthreads();
    if (t < 64) score[wgRow + t] = sAcc[t];
}

// ---------------------------------------------------------------------------
// Kernel 3: monotonic attention recurrence per batch row (two block scans).
// ---------------------------------------------------------------------------
__global__ __launch_bounds__(256) void mono_scan_kernel(
    const float* __restrict__ score, const float* __restrict__ prev_att,
    float* __restrict__ attn) {
    __shared__ float sh[256];
    const int b = blockIdx.x;
    const int t = threadIdx.x;
    const int base = b * SEQ + t * 8;

    float p[8], lpre[8];
    float run = 0.f;
    #pragma unroll
    for (int j = 0; j < 8; ++j) {
        const float s = score[base + j];
        const float pj = 1.0f / (1.0f + expf(-s));
        p[j] = pj;
        float om = 1.0f - pj;
        om = fminf(fmaxf(om, EPS), 1.0f);
        run += logf(om);
        lpre[j] = run;                           // inclusive local prefix
    }
    sh[t] = run;
    __syncthreads();
    for (int off = 1; off < 256; off <<= 1) {    // Hillis-Steele inclusive scan
        const float add = (t >= off) ? sh[t - off] : 0.0f;
        __syncthreads();
        sh[t] += add;
        __syncthreads();
    }
    const float off1 = sh[t] - run;              // exclusive thread offset

    float cp[8], r[8];
    float run2 = 0.f;
    #pragma unroll
    for (int j = 0; j < 8; ++j) {
        const float excl = off1 + (j ? lpre[j - 1] : 0.0f);
        const float c = expf(excl);              // cumprod_{i<j}(1-p)
        cp[j] = c;
        const float cc = fminf(fmaxf(c, EPS), 1.0f);
        run2 += prev_att[base + j] / cc;
        r[j] = run2;
    }
    __syncthreads();
    sh[t] = run2;
    __syncthreads();
    for (int off = 1; off < 256; off <<= 1) {
        const float add = (t >= off) ? sh[t - off] : 0.0f;
        __syncthreads();
        sh[t] += add;
        __syncthreads();
    }
    const float off2 = sh[t] - run2;
    #pragma unroll
    for (int j = 0; j < 8; ++j)
        attn[base + j] = p[j] * cp[j] * (off2 + r[j]);
}

// ---------------------------------------------------------------------------
// Kernel 4: context[b,d] = sum_s attn[b,s] * enc[b,s,d]  (bandwidth bound)
// ---------------------------------------------------------------------------
__global__ __launch_bounds__(256) void context_kernel(
    const float* __restrict__ attn, const float* __restrict__ enc,
    float* __restrict__ ctx) {
    const int b = blockIdx.x;
    const int d = blockIdx.y * 256 + threadIdx.x;
    const float* e = enc + (size_t)b * SEQ * DIM + d;
    const float* a = attn + (size_t)b * SEQ;
    float acc = 0.f;
    #pragma unroll 8
    for (int s = 0; s < SEQ; ++s)
        acc = fmaf(a[s], e[(size_t)s * DIM], acc);
    ctx[(size_t)b * DIM + d] = acc;
}

// ---------------------------------------------------------------------------
// Launch
// ---------------------------------------------------------------------------
extern "C" void kernel_launch(void* const* d_in, const int* in_sizes, int n_in,
                              void* d_out, int out_size, void* d_ws, size_t ws_size,
                              hipStream_t stream) {
    const float* dec  = (const float*)d_in[0];   // [64,1024]
    const float* enc  = (const float*)d_in[1];   // [64,2048,1024]
    const float* prev = (const float*)d_in[2];   // [64,2048]
    const float* Wa   = (const float*)d_in[3];   // [1024,1024]
    const float* Ua   = (const float*)d_in[4];   // [1024,1024]
    const float* Uab  = (const float*)d_in[5];   // [1024]
    const float* Va   = (const float*)d_in[6];   // [1024,1]

    float* out  = (float*)d_out;
    float* ctx  = out;                           // [64,1024]
    float* attn = out + BATCH * DIM;             // [64,2048]

    unsigned char* ws = (unsigned char*)d_ws;
    float* decproj        = (float*)ws;                      // 256 KB
    float* score          = (float*)(ws + 262144);           // 512 KB
    unsigned short* UaT   = (unsigned short*)(ws + 786432);  // 2 MB bf16

    ua_transpose_bf16<<<dim3(32, 32), dim3(32, 8), 0, stream>>>(Ua, UaT);
    decproj_kernel<<<BATCH, 256, 0, stream>>>(dec, Wa, decproj);

    const size_t smem = (size_t)64 * ASTRIDE * 2      // A stage
                      + (UNITS + UNITS + 64) * 4;     // dpb + va + sAcc
    fused_score_kernel<<<(BATCH * SEQ) / 64, 256, smem, stream>>>(
        enc, UaT, decproj, Uab, Va, score);

    mono_scan_kernel<<<BATCH, 256, 0, stream>>>(score, prev, attn);
    context_kernel<<<dim3(BATCH, DIM / 256), 256, 0, stream>>>(attn, enc, ctx);
}
